// _KDA_Fast_45380624449570
// MI455X (gfx1250) — matrix operand order repaired and checked by reading, not compiled, not run on hardware
//
#include <hip/hip_runtime.h>
#include <hip/hip_bf16.h>
#include <stdint.h>

// ---------------------------------------------------------------------------
// KDA forward for MI455X (gfx1250, wave32, WMMA).
//   B=2 T=2048 D=1024 H=16 DK=DV=64 R=64 C=64 KW=4   -> M = B*T = 4096 tokens
// ---------------------------------------------------------------------------

typedef __bf16 bf16;
typedef __bf16        bf16x16 __attribute__((ext_vector_type(16)));
typedef float         f32x8   __attribute__((ext_vector_type(8)));
typedef float         f32x2   __attribute__((ext_vector_type(2)));
typedef unsigned int  u32x4   __attribute__((ext_vector_type(4)));
typedef int           i32x4   __attribute__((ext_vector_type(4)));
typedef int           i32x8   __attribute__((ext_vector_type(8)));

#define USE_TDM 1

__device__ __forceinline__ float sigmoidf_(float v){ return 1.f/(1.f+__expf(-v)); }

// A fragment, 16x32 bf16, lane l<16: rows M=l, K = k0+{0..7, 16..23};
// lanes >=16: K = k0+{8..15, 24..31}.  Caller passes row pointer + k0 + (l>=16?8:0).
__device__ __forceinline__ bf16x16 ld_fragA(const bf16* p){
  union { uint4 u; bf16 h[8]; } lo, hi;
  lo.u = *(const uint4*)(p);
  hi.u = *(const uint4*)(p + 16);
  bf16x16 f;
#pragma unroll
  for (int i=0;i<8;i++){ f[i]=lo.h[i]; f[i+8]=hi.h[i]; }
  return f;
}
// B fragment, 32x16 bf16: lane l holds column n=l&15, K = k0 + (l>>4)*8 + {0..7, 16..23}.
__device__ __forceinline__ bf16x16 ld_fragB(const bf16* p){
  union { uint4 u; bf16 h[8]; } lo, hi;
  lo.u = *(const uint4*)(p);
  hi.u = *(const uint4*)(p + 16);
  bf16x16 f;
#pragma unroll
  for (int i=0;i<8;i++){ f[i]=lo.h[i]; f[i+8]=hi.h[i]; }
  return f;
}

// ---------------------------------------------------------------------------
// Generic bf16 WMMA GEMM:  Y[M,N] = act(A[M,K] * W[N,K]^T + bias[N])
// Block = 256 threads = 8 waves, each wave: 16 rows x (16*NT) cols.
// ACT: 0=none 1=silu 2=sigmoid 3=sigmoid+clip(0.1,1)
// ---------------------------------------------------------------------------
template<int NT, int ACT, bool OUTBF>
__global__ __launch_bounds__(256) void gemm_wmma(
    const bf16* __restrict__ A, const bf16* __restrict__ W,
    const float* __restrict__ bias, float* __restrict__ Yf, bf16* __restrict__ Yb,
    int M, int N, int K)
{
  (void)M;
  const int lane = threadIdx.x & 31;
  const int wv   = threadIdx.x >> 5;
  const int l15  = lane & 15;
  const int lh   = lane >> 4;
  const int rowBase = blockIdx.x * 128 + wv * 16;
  const int colBase = blockIdx.y * (16 * NT);

  f32x8 acc[NT];
#pragma unroll
  for (int t=0;t<NT;t++)
#pragma unroll
    for (int r=0;r<8;r++) acc[t][r] = 0.f;

  const bf16* arow = A + (size_t)(rowBase + l15) * K;
  for (int k0=0;k0<K;k0+=32){
    __builtin_prefetch(arow + k0 + 128, 0, 0);
    bf16x16 af = ld_fragA(arow + k0 + lh*8);
#pragma unroll
    for (int t=0;t<NT;t++){
      const bf16* wrow = W + (size_t)(colBase + t*16 + l15) * K + k0 + lh*8;
      bf16x16 bfr = ld_fragB(wrow);
      acc[t] = __builtin_amdgcn_wmma_f32_16x16x32_bf16(
          false, af, false, bfr, (short)0, acc[t], false, false);
    }
  }
  // C/D layout: element (row = r + 8*(lane>=16), col = lane&15) per 16x16 tile
#pragma unroll
  for (int t=0;t<NT;t++){
    const int col = colBase + t*16 + l15;
    const float bv = bias[col];
#pragma unroll
    for (int r=0;r<8;r++){
      const int row = rowBase + r + (lh<<3);
      float v = acc[t][r] + bv;
      if (ACT==1)      v = v * sigmoidf_(v);
      else if (ACT==2) v = sigmoidf_(v);
      else if (ACT==3) v = fmaxf(sigmoidf_(v), 0.1f);
      if (OUTBF) Yb[(size_t)row*N + col] = (bf16)v;
      else       Yf[(size_t)row*N + col] = v;
    }
  }
}

// ---------------------------------------------------------------------------
// Causal depthwise shortconv (KW=4) + SiLU, f32 in -> bf16 out. One thread/elem.
// ---------------------------------------------------------------------------
__global__ __launch_bounds__(256) void shortconv_silu(
  const float* __restrict__ x, const float* __restrict__ w,
  const float* __restrict__ bias, bf16* __restrict__ y)
{
  const size_t id = (size_t)blockIdx.x*256 + threadIdx.x;   // < 4096*1024
  const int d = (int)(id & 1023);
  const size_t bt = id >> 10;                               // b*2048 + t
  const int t = (int)(bt & 2047);
  float acc = bias[d];
#pragma unroll
  for (int j=0;j<4;j++){
    int ts = t - 3 + j;
    if (ts >= 0) acc += x[(bt - t + ts)*1024 + d] * w[d*4 + j];
  }
  y[id] = (bf16)(acc * sigmoidf_(acc));
}

__global__ __launch_bounds__(256) void cvt_bf16(
  const float* __restrict__ in, bf16* __restrict__ out, int n)
{
  int i = blockIdx.x*256 + threadIdx.x;
  if (i < n) out[i] = (bf16)in[i];
}

// Per-(token, head) L2 normalization in place: one wave per 64-wide head slice.
__global__ __launch_bounds__(256) void l2norm_heads(float* __restrict__ P, int rows)
{
  int gw   = (blockIdx.x * 256 + threadIdx.x) >> 5;
  int lane = threadIdx.x & 31;
  int row = gw >> 4, h = gw & 15;
  if (row >= rows) return;
  float* p = P + (size_t)row*1024 + h*64;
  float a = p[lane], b = p[lane+32];
  float s = a*a + b*b;
#pragma unroll
  for (int off=16; off>0; off>>=1) s += __shfl_xor(s, off, 32);
  float inv = rsqrtf(fmaxf(s, 1e-6f));
  p[lane] = a*inv; p[lane+32] = b*inv;
}

// ---------------------------------------------------------------------------
// Chunked gated delta-rule scan. One block per (b,h); 256 threads; 32 chunks of 64.
//   S[64x64] f32 state in LDS.  O = Q@S_prev + A_att@V  (f32 WMMA 16x16x4),
//   exact sequential per-timestep state update, RMSNorm + gate epilogue.
// Alpha tile loaded via Tensor Data Mover (TDM) into LDS.
// ---------------------------------------------------------------------------
__global__ __launch_bounds__(256) void chunk_scan(
  const float* __restrict__ Qp, const float* __restrict__ Kp, const float* __restrict__ Vp,
  const float* __restrict__ Ap, const float* __restrict__ Bp, const float* __restrict__ Gp,
  const float* __restrict__ rmsw, bf16* __restrict__ On)
{
  __shared__ float S[4096];     // state, persists across chunks
  __shared__ float CLA[4096];   // alpha tile -> in-place cumulative log decay
  __shared__ float ATO[4096];   // attention scores, then re-used as O tile
  __shared__ float Bsh[64];
  __shared__ float GS[64];
  __shared__ float PK[256];
  __shared__ float RV[64];

  const int tid = threadIdx.x;
  const int bh = blockIdx.x;
  const int b = bh >> 4, h = bh & 15;
  const int lane = tid & 31, wv = tid >> 5;
  const int l15 = lane & 15, lh = lane >> 4;

#pragma unroll
  for (int r=0;r<16;r++) S[tid + 256*r] = 0.f;
  __syncthreads();

  for (int n=0;n<32;n++){
    const size_t rowbase = (size_t)b*2048 + (size_t)n*64;
    const float* Qg = Qp + rowbase*1024 + h*64;
    const float* Kg = Kp + rowbase*1024 + h*64;
    const float* Vg = Vp + rowbase*1024 + h*64;
    const float* Ag = Ap + rowbase*1024 + h*64;

    // ---- load alpha tile (64x64, row stride 1024 f32) into CLA via TDM ----
#if USE_TDM && __has_builtin(__builtin_amdgcn_tensor_load_to_lds)
    if (tid < 32){
      u32x4 g0; i32x8 g1;
      i32x4 g2 = {0,0,0,0}, g3 = {0,0,0,0};
      unsigned long long ga = (unsigned long long)(uintptr_t)Ag;
      unsigned ldsa = (unsigned)(uintptr_t)(void*)&CLA[0];   // low 32b = LDS offset
      g0[0] = 1u;                                            // count=1, no gather
      g0[1] = ldsa;                                          // lds_addr
      g0[2] = (unsigned)(ga & 0xffffffffull);                // global_addr lo
      g0[3] = (unsigned)((ga >> 32) & 0x1ffffffull) | (2u << 30); // hi + type=2
      g1[0] = (int)(2u << 16);    // data_size=4B, wg_mask=0
      g1[1] = (int)(64u << 16);   // tensor_dim0 = 64
      g1[2] = (int)(64u << 16);   // tensor_dim1 = 64
      g1[3] = (int)(64u << 16);   // tile_dim0 = 64
      g1[4] = (int)(64u);         // tile_dim1 = 64, tile_dim2 = 0 (2D)
      g1[5] = 1024;               // tensor_dim0_stride = 1024
      g1[6] = 0; g1[7] = 0;
#if defined(__clang_major__) && (__clang_major__ >= 23)
      i32x8 g4 = {0,0,0,0,0,0,0,0};
      __builtin_amdgcn_tensor_load_to_lds(g0, g1, g2, g3, g4, 0);
#else
      __builtin_amdgcn_tensor_load_to_lds(g0, g1, g2, g3, 0);
#endif
      __builtin_amdgcn_s_wait_tensorcnt(0);
    }
#else
#pragma unroll
    for (int r=0;r<16;r++){
      int idx = tid + 256*r;
      CLA[idx] = Ag[(size_t)(idx>>6)*1024 + (idx & 63)];
    }
#endif
    if (tid < 64){
      Bsh[tid] = fminf(Bp[(rowbase + tid)*16 + h], 1.0f);
      GS[tid]  = Gp[(rowbase + tid)*16 + h];
    }
    __syncthreads();

    // ---- in-place cumulative log(alpha) per feature d ----
    if (tid < 64){
      float accl = 0.f;
      for (int c=0;c<64;c++){
        accl += __logf(fmaxf(CLA[c*64 + tid], 1e-7f));
        CLA[c*64 + tid] = accl;
      }
    }
    __syncthreads();

    // ---- intra-chunk decay-weighted scores: A[i][j] = b_j * sum_d q*k*exp(cl_i-cl_j) ----
#pragma unroll
    for (int r=0;r<16;r++){
      int idx = tid + 256*r;
      int i = idx >> 6, j = idx & 63;
      float v = 0.f;
      if (i >= j){
        const float* qi = Qg + (size_t)i*1024;
        const float* kj = Kg + (size_t)j*1024;
#pragma unroll 8
        for (int d=0; d<64; d++)
          v += qi[d]*kj[d]*__expf(CLA[i*64+d] - CLA[j*64+d]);
        v *= Bsh[j];
      }
      ATO[idx] = v;
    }
    __syncthreads();

    // ---- O = Q @ S + A_att @ V  via f32 WMMA 16x16x4 (2 tiles per wave) ----
    f32x8 accT[2];
#pragma unroll
    for (int s=0;s<2;s++){
      const int tile = wv*2 + s, ti = tile >> 2, tj = tile & 3;
      f32x8 acc;
#pragma unroll
      for (int r=0;r<8;r++) acc[r]=0.f;
      const float* qrow = Qg + (size_t)(ti*16 + l15)*1024;
      for (int d0=0; d0<64; d0+=4){
        f32x2 a;  a[0]  = qrow[d0 + 2*lh];  a[1]  = qrow[d0 + 2*lh + 1];
        f32x2 bb; bb[0] = S[(d0 + 2*lh)*64 + tj*16 + l15];
                  bb[1] = S[(d0 + 2*lh + 1)*64 + tj*16 + l15];
        acc = __builtin_amdgcn_wmma_f32_16x16x4_f32(false, a, false, bb, (short)0, acc, false, false);
      }
      const float* atrow = ATO + (ti*16 + l15)*64;
      for (int j0=0; j0<64; j0+=4){
        f32x2 a;  a[0]  = atrow[j0 + 2*lh];  a[1] = atrow[j0 + 2*lh + 1];
        f32x2 bb; bb[0] = Vg[(size_t)(j0 + 2*lh)*1024 + tj*16 + l15];
                  bb[1] = Vg[(size_t)(j0 + 2*lh + 1)*1024 + tj*16 + l15];
        acc = __builtin_amdgcn_wmma_f32_16x16x4_f32(false, a, false, bb, (short)0, acc, false, false);
      }
      accT[s] = acc;
    }
    __syncthreads();          // all reads of ATO done; now reuse it as O tile
#pragma unroll
    for (int s=0;s<2;s++){
      const int tile = wv*2 + s, ti = tile >> 2, tj = tile & 3;
#pragma unroll
      for (int r=0;r<8;r++)
        ATO[(ti*16 + r + (lh<<3))*64 + tj*16 + l15] = accT[s][r];
    }
    __syncthreads();

    // ---- headwise RMSNorm * rms_w * gate, store bf16 [4096, 1024] ----
    if (tid < 64){
      float ss = 0.f;
      for (int e=0;e<64;e++){ float o = ATO[tid*64+e]; ss += o*o; }
      RV[tid] = rsqrtf(ss*(1.f/64.f) + 1e-6f) * GS[tid];
    }
    __syncthreads();
#pragma unroll
    for (int r=0;r<16;r++){
      int idx = tid + 256*r;
      int c = idx >> 6, e = idx & 63;
      float val = ATO[idx] * RV[c] * rmsw[h*64 + e];
      On[(rowbase + c)*1024 + h*64 + e] = (bf16)val;
    }
    __syncthreads();

    // ---- exact sequential delta-rule state update over the chunk ----
    const int e = tid & 63, dg = tid >> 6;   // thread owns column e, 16 rows
    for (int t=0;t<64;t++){
      const float ve = Vg[(size_t)t*1024 + e];
      const float bt = Bsh[t];
      const float* krow = Kg + (size_t)t*1024;
      const float* arow = Ag + (size_t)t*1024;
      float ar[16], kr[16];
      float partial = 0.f;
#pragma unroll
      for (int dd=0;dd<16;dd++){
        const int d = dg*16 + dd;
        float a = arow[d], k = krow[d];
        ar[dd]=a; kr[dd]=k;
        partial += a*k*S[d*64 + e];          // k^T (a*S)
      }
      PK[dg*64 + e] = partial;
      __syncthreads();
      const float kts = PK[e] + PK[64+e] + PK[128+e] + PK[192+e];
      const float coeff = bt * (ve - kts);
#pragma unroll
      for (int dd=0;dd<16;dd++){
        const int d = dg*16 + dd;
        S[d*64 + e] = ar[dd]*S[d*64 + e] + kr[dd]*coeff;
      }
      __syncthreads();
    }
  }
}

// ---------------------------------------------------------------------------
extern "C" void kernel_launch(void* const* d_in, const int* in_sizes, int n_in,
                              void* d_out, int out_size, void* d_ws, size_t ws_size,
                              hipStream_t stream)
{
  (void)in_sizes; (void)n_in; (void)out_size; (void)ws_size;
  const float* x       = (const float*)d_in[0];
  const float* wq_conv = (const float*)d_in[1];  const float* bq_conv = (const float*)d_in[2];
  const float* wk_conv = (const float*)d_in[3];  const float* bk_conv = (const float*)d_in[4];
  const float* wv_conv = (const float*)d_in[5];  const float* bv_conv = (const float*)d_in[6];
  const float* Wq = (const float*)d_in[7];   const float* bq = (const float*)d_in[8];
  const float* Wk = (const float*)d_in[9];   const float* bk = (const float*)d_in[10];
  const float* Wv = (const float*)d_in[11];  const float* bv = (const float*)d_in[12];
  const float* Wad = (const float*)d_in[13]; const float* bad = (const float*)d_in[14];
  const float* Wau = (const float*)d_in[15]; const float* bau = (const float*)d_in[16];
  const float* Wbe = (const float*)d_in[17]; const float* bbe = (const float*)d_in[18];
  const float* rms_w = (const float*)d_in[19];
  const float* Wgd = (const float*)d_in[20]; const float* bgd = (const float*)d_in[21];
  const float* Wgu = (const float*)d_in[22]; const float* bgu = (const float*)d_in[23];
  const float* Wo  = (const float*)d_in[24]; const float* bo  = (const float*)d_in[25];

  const int M = 4096;                 // B*T tokens
  char* ws = (char*)d_ws;
  size_t off = 0;
  auto alloc = [&](size_t bytes)->void* {
    void* p = ws + off;
    off += (bytes + 255) & ~(size_t)255;
    return p;
  };
  bf16* xb   = (bf16*)alloc((size_t)M*1024*2);
  bf16* cq   = (bf16*)alloc((size_t)M*1024*2);
  bf16* ck   = (bf16*)alloc((size_t)M*1024*2);
  bf16* cv   = (bf16*)alloc((size_t)M*1024*2);
  bf16* Wqb  = (bf16*)alloc((size_t)1024*1024*2);
  bf16* Wkb  = (bf16*)alloc((size_t)1024*1024*2);
  bf16* Wvb  = (bf16*)alloc((size_t)1024*1024*2);
  bf16* Wob  = (bf16*)alloc((size_t)1024*1024*2);
  bf16* Wadb = (bf16*)alloc((size_t)64*1024*2);
  bf16* Waub = (bf16*)alloc((size_t)1024*64*2);
  bf16* Wbeb = (bf16*)alloc((size_t)16*1024*2);
  bf16* Wgdb = (bf16*)alloc((size_t)64*1024*2);
  bf16* Wgub = (bf16*)alloc((size_t)16*64*2);
  float* Qp  = (float*)alloc((size_t)M*1024*4);
  float* Kp  = (float*)alloc((size_t)M*1024*4);
  float* Vp  = (float*)alloc((size_t)M*1024*4);
  float* Al  = (float*)alloc((size_t)M*1024*4);
  float* Bt  = (float*)alloc((size_t)M*16*4);
  float* Gt  = (float*)alloc((size_t)M*16*4);
  bf16* t1b  = (bf16*)alloc((size_t)M*64*2);
  bf16* t2b  = (bf16*)alloc((size_t)M*64*2);
  bf16* On   = (bf16*)alloc((size_t)M*1024*2);

  const int ELTS = M*1024;            // 4,194,304

  // 1) raw x -> bf16 (for gate GEMMs)
  cvt_bf16<<<ELTS/256, 256, 0, stream>>>(x, xb, ELTS);
  // 2) shortconv + SiLU -> bf16 activations for Q/K/V projections
  shortconv_silu<<<ELTS/256, 256, 0, stream>>>(x, wq_conv, bq_conv, cq);
  shortconv_silu<<<ELTS/256, 256, 0, stream>>>(x, wk_conv, bk_conv, ck);
  shortconv_silu<<<ELTS/256, 256, 0, stream>>>(x, wv_conv, bv_conv, cv);
  // 3) weights -> bf16 (resident in 192MB L2 afterwards)
  cvt_bf16<<<(1024*1024)/256, 256, 0, stream>>>(Wq, Wqb, 1024*1024);
  cvt_bf16<<<(1024*1024)/256, 256, 0, stream>>>(Wk, Wkb, 1024*1024);
  cvt_bf16<<<(1024*1024)/256, 256, 0, stream>>>(Wv, Wvb, 1024*1024);
  cvt_bf16<<<(1024*1024)/256, 256, 0, stream>>>(Wo, Wob, 1024*1024);
  cvt_bf16<<<(64*1024)/256, 256, 0, stream>>>(Wad, Wadb, 64*1024);
  cvt_bf16<<<(64*1024)/256, 256, 0, stream>>>(Wau, Waub, 64*1024);
  cvt_bf16<<<(16*1024)/256, 256, 0, stream>>>(Wbe, Wbeb, 16*1024);
  cvt_bf16<<<(64*1024)/256, 256, 0, stream>>>(Wgd, Wgdb, 64*1024);
  cvt_bf16<<<4, 256, 0, stream>>>(Wgu, Wgub, 16*64);

  // 4) WMMA GEMMs
  gemm_wmma<4,0,false><<<dim3(32,16), 256, 0, stream>>>(cq, Wqb, bq, Qp, nullptr, M, 1024, 1024);
  gemm_wmma<4,0,false><<<dim3(32,16), 256, 0, stream>>>(ck, Wkb, bk, Kp, nullptr, M, 1024, 1024);
  gemm_wmma<4,0,false><<<dim3(32,16), 256, 0, stream>>>(cv, Wvb, bv, Vp, nullptr, M, 1024, 1024);
  gemm_wmma<4,1,true ><<<dim3(32, 1), 256, 0, stream>>>(xb, Wadb, bad, nullptr, t1b, M, 64, 1024);
  gemm_wmma<4,3,false><<<dim3(32,16), 256, 0, stream>>>(t1b, Waub, bau, Al, nullptr, M, 1024, 64);
  gemm_wmma<1,2,false><<<dim3(32, 1), 256, 0, stream>>>(xb, Wbeb, bbe, Bt, nullptr, M, 16, 1024);
  gemm_wmma<4,1,true ><<<dim3(32, 1), 256, 0, stream>>>(xb, Wgdb, bgd, nullptr, t2b, M, 64, 1024);
  gemm_wmma<1,2,false><<<dim3(32, 1), 256, 0, stream>>>(t2b, Wgub, bgu, Gt, nullptr, M, 16, 64);

  // 5) per-head L2 norm of Q, K (in place)
  l2norm_heads<<<(M*16)/8, 256, 0, stream>>>(Qp, M);
  l2norm_heads<<<(M*16)/8, 256, 0, stream>>>(Kp, M);

  // 6) chunked gated delta-rule scan (one block per head-batch)
  chunk_scan<<<32, 256, 0, stream>>>(Qp, Kp, Vp, Al, Bt, Gt, rms_w, On);

  // 7) output projection
  gemm_wmma<4,0,false><<<dim3(32,16), 256, 0, stream>>>(On, Wob, bo, (float*)d_out, nullptr, M, 1024, 1024);
}
